// MatryoshkaSAE_35931696399064
// MI455X (gfx1250) — matrix operand order repaired and checked by reading, not compiled, not run on hardware
//
#include <hip/hip_runtime.h>

#define D_IN   768
#define D_SAE  16384
#define BATCH  4096
#define TOTALK 262144u
#define NTOT   (1u << 26)

typedef __attribute__((ext_vector_type(16))) __bf16          bf16x16;
typedef __attribute__((ext_vector_type(8)))  float           floatx8;
typedef __attribute__((ext_vector_type(8)))  unsigned short  ushort8;
typedef __attribute__((ext_vector_type(16))) unsigned short  ushort16;

__device__ __forceinline__ unsigned short f32_to_bf16_rne(float f) {
  unsigned u = __float_as_uint(f);
  unsigned r = u + 0x7FFFu + ((u >> 16) & 1u);
  return (unsigned short)(r >> 16);
}
__device__ __forceinline__ float bf16_bits_to_f32(unsigned short h) {
  return __uint_as_float(((unsigned)h) << 16);
}

__device__ __forceinline__ bf16x16 load_frag(const unsigned short* p0,
                                             const unsigned short* p1) {
  ushort8 lo = *(const ushort8*)p0;   // ds_load_b128
  ushort8 hi = *(const ushort8*)p1;   // ds_load_b128
  ushort16 u;
#pragma unroll
  for (int i = 0; i < 8; i++) { u[i] = lo[i]; u[i + 8] = hi[i]; }
  return __builtin_bit_cast(bf16x16, u);
}

// ---------------------------------------------------------------------------
// Encoder GEMM: pre = relu((x - b_dec) @ W_enc + b_enc) -> z (written dense)
// 128x128 block tile, BK=32, 8 waves, each wave owns 16 rows x 128 cols.
// bf16 hi/lo split: 3 WMMAs per 16x16x32 step => near-f32 accuracy.
// Register double-buffer: next k-step's global loads are issued right after
// the staging barrier into the (already-consumed) staging registers, so VMEM
// latency overlaps the 24 WMMAs of the current step.
// Also builds the level-0 (bits 31:24) radix histogram in the epilogue.
// ---------------------------------------------------------------------------
#define BM 128
#define BN 128
#define BK 32
#define LDT 40   // padded row stride in ushorts (80B, 16B aligned)

__device__ __forceinline__ void gload_tiles(const float* __restrict__ x,
                                            const float* __restrict__ W_enc,
                                            const float* __restrict__ b_dec,
                                            int m0, int n0, int k0, int tid,
                                            float4 ra[4], float4 rb[4]) {
#pragma unroll
  for (int i = 0; i < 4; i++) {              // A tile: 128x32 f32
    int e4 = tid + i * 256;
    int r = e4 >> 3, c4 = (e4 & 7) * 4;
    float4 v = *(const float4*)(x + (size_t)(m0 + r) * D_IN + k0 + c4);
    float4 b = *(const float4*)(b_dec + k0 + c4);
    v.x -= b.x; v.y -= b.y; v.z -= b.z; v.w -= b.w;
    ra[i] = v;
  }
#pragma unroll
  for (int i = 0; i < 4; i++) {              // B tile: 32x128 f32
    int e4 = tid + i * 256;
    int kk = e4 >> 5, nn4 = (e4 & 31) * 4;
    rb[i] = *(const float4*)(W_enc + (size_t)(k0 + kk) * D_SAE + n0 + nn4);
  }
}

__global__ __launch_bounds__(256)
void sae_encode_wmma(const float* __restrict__ x, const float* __restrict__ W_enc,
                     const float* __restrict__ b_enc, const float* __restrict__ b_dec,
                     float* __restrict__ z, unsigned* __restrict__ hist) {
  __shared__ __attribute__((aligned(16))) unsigned short Ah[BM][LDT];
  __shared__ __attribute__((aligned(16))) unsigned short Al[BM][LDT];
  __shared__ __attribute__((aligned(16))) unsigned short Bh[BN][LDT];
  __shared__ __attribute__((aligned(16))) unsigned short Bl[BN][LDT];
  __shared__ unsigned lhist[256];

  const int tid  = threadIdx.x;
  const int wave = tid >> 5;
  const int lane = tid & 31;
  const int m0 = blockIdx.y * BM;
  const int n0 = blockIdx.x * BN;

  lhist[tid] = 0;

  const floatx8 zerov = {0.f, 0.f, 0.f, 0.f, 0.f, 0.f, 0.f, 0.f};
  floatx8 acc[8];
#pragma unroll
  for (int t = 0; t < 8; t++) acc[t] = zerov;

  float4 ra[4], rb[4];
  gload_tiles(x, W_enc, b_dec, m0, n0, 0, tid, ra, rb);

  for (int k0 = 0; k0 < D_IN; k0 += BK) {
    // ---- stage current tiles from registers into LDS (hi/lo bf16 split) ----
#pragma unroll
    for (int i = 0; i < 4; i++) {
      int e4 = tid + i * 256;
      int r = e4 >> 3, c4 = (e4 & 7) * 4;
      float vv[4] = {ra[i].x, ra[i].y, ra[i].z, ra[i].w};
      unsigned long long hp = 0ull, lp = 0ull;
#pragma unroll
      for (int j = 0; j < 4; j++) {
        unsigned short h = f32_to_bf16_rne(vv[j]);
        unsigned short l = f32_to_bf16_rne(vv[j] - bf16_bits_to_f32(h));
        hp |= (unsigned long long)h << (16 * j);
        lp |= (unsigned long long)l << (16 * j);
      }
      *(unsigned long long*)&Ah[r][c4] = hp;   // ds_store_b64
      *(unsigned long long*)&Al[r][c4] = lp;
    }
#pragma unroll
    for (int i = 0; i < 4; i++) {              // transpose into N-major [128][32]
      int e4 = tid + i * 256;
      int kk = e4 >> 5, nn4 = (e4 & 31) * 4;
      float vv[4] = {rb[i].x, rb[i].y, rb[i].z, rb[i].w};
#pragma unroll
      for (int j = 0; j < 4; j++) {
        unsigned short h = f32_to_bf16_rne(vv[j]);
        Bh[nn4 + j][kk] = h;
        Bl[nn4 + j][kk] = f32_to_bf16_rne(vv[j] - bf16_bits_to_f32(h));
      }
    }
    __syncthreads();

    // ---- issue next k-step's global loads; they overlap the WMMAs below ----
    const bool notlast = (k0 + BK < D_IN);
    if (notlast) gload_tiles(x, W_enc, b_dec, m0, n0, k0 + BK, tid, ra, rb);

    // ---- fragments + WMMA ----
    const int arow  = wave * 16 + (lane & 15);
    const int abase = (lane >> 4) * 8;             // A: lanes 16-31 -> K+8
    bf16x16 ahf = load_frag(&Ah[arow][abase], &Ah[arow][16 + abase]);
    bf16x16 alf = load_frag(&Al[arow][abase], &Al[arow][16 + abase]);
#pragma unroll
    for (int t = 0; t < 8; t++) {
      const int brow  = t * 16 + (lane & 15);
      const int bbase = (lane >> 4) * 8;           // B: lanes 16-31 -> K+8
      bf16x16 bhf = load_frag(&Bh[brow][bbase], &Bh[brow][bbase + 16]);
      bf16x16 blf = load_frag(&Bl[brow][bbase], &Bl[brow][bbase + 16]);
      acc[t] = __builtin_amdgcn_wmma_f32_16x16x32_bf16(false, ahf, false, bhf,
                                                       (short)0, acc[t], false, false);
      acc[t] = __builtin_amdgcn_wmma_f32_16x16x32_bf16(false, ahf, false, blf,
                                                       (short)0, acc[t], false, false);
      acc[t] = __builtin_amdgcn_wmma_f32_16x16x32_bf16(false, alf, false, bhf,
                                                       (short)0, acc[t], false, false);
    }
    __syncthreads();
  }

  // epilogue: add b_enc, relu, store z, accumulate level-0 histogram
  const int mrow0 = m0 + wave * 16 + ((lane >> 4) << 3);
#pragma unroll
  for (int t = 0; t < 8; t++) {
    const int n = n0 + t * 16 + (lane & 15);
    const float be = b_enc[n];
#pragma unroll
    for (int r = 0; r < 8; r++) {
      float v = acc[t][r] + be;
      v = v > 0.f ? v : 0.f;
      z[(size_t)(mrow0 + r) * D_SAE + n] = v;
      atomicAdd(&lhist[__float_as_uint(v) >> 24], 1u);
    }
  }
  __syncthreads();
  atomicAdd(&hist[tid], lhist[tid]);
}

// ---------------------------------------------------------------------------
// Radix select over float bit patterns (monotonic for the >=0 relu outputs)
// ws layout: state[0]=prefix, state[1]=k_remaining, state[2..257]=hist[256]
// ---------------------------------------------------------------------------
__global__ void radix_init(unsigned* state) {
  int t = threadIdx.x;
  if (t == 0) { state[0] = 0u; state[1] = TOTALK; }
  state[2 + t] = 0u;
}

__global__ __launch_bounds__(256)
void radix_hist(const float* __restrict__ z, unsigned* __restrict__ state, int shift) {
  __shared__ unsigned lhist[256];
  lhist[threadIdx.x] = 0u;
  __syncthreads();
  const unsigned pref_hi = state[0] >> (shift + 8);   // shift in {16,8,0}
  const float4* z4 = (const float4*)z;
  unsigned i = blockIdx.x * 256u + threadIdx.x;
  const unsigned stride = gridDim.x * 256u;
  for (; i < (NTOT >> 2); i += stride) {
    float4 v = z4[i];
    unsigned u0 = __float_as_uint(v.x), u1 = __float_as_uint(v.y);
    unsigned u2 = __float_as_uint(v.z), u3 = __float_as_uint(v.w);
    if ((u0 >> (shift + 8)) == pref_hi) atomicAdd(&lhist[(u0 >> shift) & 255u], 1u);
    if ((u1 >> (shift + 8)) == pref_hi) atomicAdd(&lhist[(u1 >> shift) & 255u], 1u);
    if ((u2 >> (shift + 8)) == pref_hi) atomicAdd(&lhist[(u2 >> shift) & 255u], 1u);
    if ((u3 >> (shift + 8)) == pref_hi) atomicAdd(&lhist[(u3 >> shift) & 255u], 1u);
  }
  __syncthreads();
  atomicAdd(&state[2 + threadIdx.x], lhist[threadIdx.x]);
}

__global__ __launch_bounds__(256)
void radix_select(unsigned* state, int shift) {
  __shared__ unsigned h[256];
  const int t = threadIdx.x;
  const unsigned krem = state[1];
  const unsigned cnt = state[2 + t];
  h[t] = cnt;
  __syncthreads();
  unsigned gt = 0;
  for (int j = t + 1; j < 256; j++) gt += h[j];
  if (gt < krem && gt + cnt >= krem) {   // exactly one bin satisfies this
    state[0] |= ((unsigned)t) << shift;
    state[1] = krem - gt;
  }
  state[2 + t] = 0u;                     // reset hist for next pass
}

__global__ __launch_bounds__(256)
void topk_apply(float* __restrict__ z, const unsigned* __restrict__ state) {
  const unsigned T = state[0];
  float4* z4 = (float4*)z;
  unsigned i = blockIdx.x * 256u + threadIdx.x;
  const unsigned stride = gridDim.x * 256u;
  for (; i < (NTOT >> 2); i += stride) {
    float4 v = z4[i];
    v.x = (__float_as_uint(v.x) >= T) ? v.x : 0.f;
    v.y = (__float_as_uint(v.y) >= T) ? v.y : 0.f;
    v.z = (__float_as_uint(v.z) >= T) ? v.z : 0.f;
    v.w = (__float_as_uint(v.w) >= T) ? v.w : 0.f;
    z4[i] = v;
  }
}

// ---------------------------------------------------------------------------
// Sparse decode: one block per batch row. The z-row scan is a double-buffered
// global_load_async_to_lds_b32 pipeline (ASYNCcnt completes in order, so
// s_wait_asynccnt 1 covers the current chunk while the next is in flight).
// Deterministic ballot-ordered compaction, then 3 cols per thread (768=3*256),
// with global_prefetch on the next gathered W_dec row.
// ---------------------------------------------------------------------------
__global__ __launch_bounds__(256)
void sae_decode(const float* __restrict__ z, const float* __restrict__ W_dec,
                const float* __restrict__ b_dec, float* __restrict__ xhat) {
  __shared__ unsigned short sidx[D_SAE];
  __shared__ __attribute__((aligned(16))) float zb[2][256];
  __shared__ unsigned wbase[8];
  __shared__ unsigned runtot;
  const int tid  = threadIdx.x;
  const int wave = tid >> 5;
  const int lane = tid & 31;
  const int m = blockIdx.x;
  const float* zr = z + (size_t)m * D_SAE;

  // LDS byte offset of zb[0][tid] (addrspace(3) ptrtoint)
  typedef __attribute__((address_space(3))) char lds_char;
  const unsigned zb0 = (unsigned)(unsigned long long)(lds_char*)(&zb[0][0])
                     + (unsigned)tid * 4u;

  if (tid == 0) runtot = 0u;

  // prologue: async-stage chunk 0 into zb[0]
  asm volatile("global_load_async_to_lds_b32 %0, %1, off"
               :: "v"(zb0), "v"(zr + tid) : "memory");
  __syncthreads();

  for (int c0 = 0; c0 < D_SAE; c0 += 256) {
    const int buf = (c0 >> 8) & 1;
    if (c0 + 256 < D_SAE) {
      const unsigned dst = zb0 + (unsigned)((buf ^ 1) << 10);
      asm volatile("global_load_async_to_lds_b32 %0, %1, off"
                   :: "v"(dst), "v"(zr + c0 + 256 + tid) : "memory");
      asm volatile("s_wait_asynccnt 0x1" ::: "memory");
    } else {
      asm volatile("s_wait_asynccnt 0x0" ::: "memory");
    }
    float v = zb[buf][tid];
    bool p = (v != 0.f);
    unsigned mask = (unsigned)__ballot(p);                // wave32
    unsigned rank = __popc(mask & ((1u << lane) - 1u));
    if (lane == 0) wbase[wave] = (unsigned)__popc(mask);
    __syncthreads();
    if (tid == 0) {
      unsigned b = runtot;
      for (int w = 0; w < 8; w++) { unsigned c = wbase[w]; wbase[w] = b; b += c; }
      runtot = b;
    }
    __syncthreads();
    if (p) sidx[wbase[wave] + rank] = (unsigned short)(c0 + tid);
    __syncthreads();
  }

  const unsigned nnz = runtot;
  float a0 = 0.f, a1 = 0.f, a2 = 0.f;
  for (unsigned pos = 0; pos < nnz; pos++) {
    const int j = sidx[pos];
    if (pos + 1 < nnz)   // global_prefetch_b8 of next gathered row
      __builtin_prefetch(W_dec + (size_t)sidx[pos + 1] * D_IN + tid, 0, 0);
    const float v = zr[j];
    const float* w = W_dec + (size_t)j * D_IN;   // L2-resident (50MB << 192MB)
    a0 += v * w[tid];
    a1 += v * w[tid + 256];
    a2 += v * w[tid + 512];
  }
  xhat[(size_t)m * D_IN + tid]       = a0 + b_dec[tid];
  xhat[(size_t)m * D_IN + tid + 256] = a1 + b_dec[tid + 256];
  xhat[(size_t)m * D_IN + tid + 512] = a2 + b_dec[tid + 512];
}

// ---------------------------------------------------------------------------
extern "C" void kernel_launch(void* const* d_in, const int* in_sizes, int n_in,
                              void* d_out, int out_size, void* d_ws, size_t ws_size,
                              hipStream_t stream) {
  (void)in_sizes; (void)n_in; (void)out_size; (void)ws_size;
  const float* x     = (const float*)d_in[0];
  const float* W_enc = (const float*)d_in[1];
  const float* b_enc = (const float*)d_in[2];
  const float* W_dec = (const float*)d_in[3];
  const float* b_dec = (const float*)d_in[4];

  float* xhat = (float*)d_out;                        // [4096, 768]
  float* z    = xhat + (size_t)BATCH * D_IN;          // [4096, 16384]
  unsigned* state = (unsigned*)d_ws;                  // 258 u32 of scratch

  radix_init<<<1, 256, 0, stream>>>(state);

  dim3 grid(D_SAE / BN, BATCH / BM);                  // (128, 32)
  sae_encode_wmma<<<grid, 256, 0, stream>>>(x, W_enc, b_enc, b_dec, z, state + 2);

  radix_select<<<1, 256, 0, stream>>>(state, 24);     // hist level 0 built in GEMM
  radix_hist  <<<2048, 256, 0, stream>>>(z, state, 16);
  radix_select<<<1, 256, 0, stream>>>(state, 16);
  radix_hist  <<<2048, 256, 0, stream>>>(z, state, 8);
  radix_select<<<1, 256, 0, stream>>>(state, 8);
  radix_hist  <<<2048, 256, 0, stream>>>(z, state, 0);
  radix_select<<<1, 256, 0, stream>>>(state, 0);

  topk_apply<<<2048, 256, 0, stream>>>(z, state);
  sae_decode<<<BATCH, 256, 0, stream>>>(z, W_dec, b_dec, xhat);
}
